// GATLayerv2_82772609728559
// MI455X (gfx1250) — matrix operand order repaired and checked by reading, not compiled, not run on hardware
//
#include <hip/hip_runtime.h>
#include <hip/hip_bf16.h>

// GAT layer for MI455X (gfx1250, wave32).
// Pass A uses v_wmma_f32_16x16x32_f16 to batch 16 edge logits per wave.

typedef __attribute__((ext_vector_type(16))) _Float16 v16h;
typedef __attribute__((ext_vector_type(8)))  float    v8f;

#define D_DIM 64
#define NEG_SLOPE 0.1f

// ---- monotonic encoding for float atomic-max via u32 atomicMax ----
__device__ __forceinline__ unsigned f2mono(float f) {
    unsigned u = __float_as_uint(f);
    return (u & 0x80000000u) ? ~u : (u | 0x80000000u);
}
__device__ __forceinline__ float mono2f(unsigned u) {
    return (u & 0x80000000u) ? __uint_as_float(u & 0x7FFFFFFFu)
                             : __uint_as_float(~u);
}

// leaky(x) == max(x, 0.1*x) for slope < 1: one v_mul + one v_max.
__device__ __forceinline__ float leaky(float x) {
    return fmaxf(x, NEG_SLOPE * x);
}

// ---------------------------------------------------------------------------
// Kernel 0: init out[N*64]=0, smax[N]=0 (encodes -inf), denom[N]=0
// ---------------------------------------------------------------------------
__global__ void gat_init_kernel(float* __restrict__ out,
                                unsigned* __restrict__ smax,
                                float* __restrict__ denom,
                                int n_nodes) {
    int i = blockIdx.x * blockDim.x + threadIdx.x;
    int total = n_nodes * D_DIM;
    if (i < total) out[i] = 0.0f;
    if (i < n_nodes) { smax[i] = 0u; denom[i] = 0.0f; }
}

// ---------------------------------------------------------------------------
// Kernel A: edge logits via WMMA. One wave32 handles 16 edges, K=64 in two
// 16x16x32 f16 WMMA steps. Column 0 of the f32 accumulator = the 16 logits.
// Also does atomicMax into smax[dst] (monotonic u32) and stores e to elog.
// ---------------------------------------------------------------------------
__global__ void gat_logits_wmma_kernel(const float* __restrict__ h,
                                       const float* __restrict__ aw,
                                       const int* __restrict__ src,
                                       const int* __restrict__ dst,
                                       float* __restrict__ elog,
                                       unsigned* __restrict__ smax,
                                       int n_edges) {
    const int lane   = threadIdx.x & 31;
    const int waveId = threadIdx.x >> 5;
    const int wavesPerBlock = blockDim.x >> 5;
    const int ebase = (blockIdx.x * wavesPerBlock + waveId) * 16;
    if (ebase >= n_edges) return;               // wave-uniform: EXEC stays all-ones

    // This lane sources A-matrix row (lane & 15); two lanes per edge (K halves).
    const int row  = lane & 15;
    int e = ebase + row;
    if (e >= n_edges) e = n_edges - 1;          // clamp loads only (tail safety)
    const int si = src[e];
    const int di = dst[e];
    const float4* __restrict__ hs4 = (const float4*)(h + (long long)si * D_DIM);
    const float4* __restrict__ hd4 = (const float4*)(h + (long long)di * D_DIM);

    const int khalf = (lane >> 4) << 3;         // lanes 0-15 -> 0, lanes 16-31 -> 8
    const int kbcol = (lane >> 4) << 3;         // B-matrix K base offset: 0 or 8

    v8f c = {};
#pragma unroll
    for (int chunk = 0; chunk < 2; ++chunk) {
        const int kbase = chunk * 32;

        // A fragment: 16-bit A 16x32 layout (ISA 7.12.2).
        // elements 0..7  <-> K = kbase + khalf + j
        // elements 8..15 <-> K = kbase + 16 + khalf + j
        const int i0 = (kbase + khalf) >> 2;        // float4 index of first run
        const int i1 = (kbase + 16 + khalf) >> 2;   // float4 index of second run
        float4 s0 = hs4[i0],     s1 = hs4[i0 + 1];
        float4 d0 = hd4[i0],     d1 = hd4[i0 + 1];
        float4 s2 = hs4[i1],     s3 = hs4[i1 + 1];
        float4 d2 = hd4[i1],     d3 = hd4[i1 + 1];

        v16h af;
        af[0]  = (_Float16)leaky(s0.x + d0.x);
        af[1]  = (_Float16)leaky(s0.y + d0.y);
        af[2]  = (_Float16)leaky(s0.z + d0.z);
        af[3]  = (_Float16)leaky(s0.w + d0.w);
        af[4]  = (_Float16)leaky(s1.x + d1.x);
        af[5]  = (_Float16)leaky(s1.y + d1.y);
        af[6]  = (_Float16)leaky(s1.z + d1.z);
        af[7]  = (_Float16)leaky(s1.w + d1.w);
        af[8]  = (_Float16)leaky(s2.x + d2.x);
        af[9]  = (_Float16)leaky(s2.y + d2.y);
        af[10] = (_Float16)leaky(s2.z + d2.z);
        af[11] = (_Float16)leaky(s2.w + d2.w);
        af[12] = (_Float16)leaky(s3.x + d3.x);
        af[13] = (_Float16)leaky(s3.y + d3.y);
        af[14] = (_Float16)leaky(s3.z + d3.z);
        af[15] = (_Float16)leaky(s3.w + d3.w);

        // B fragment: 32x16, column-replicated a_w so every output column = e.
        // elements 0..7 <-> K = kbase + kbcol + j, elements 8..15 <-> K = kbase + 16 + kbcol + j.
        const float4* aw4 = (const float4*)(aw + kbase + kbcol);
        float4 w0 = aw4[0], w1 = aw4[1], w2 = aw4[4], w3 = aw4[5];
        v16h bf;
        bf[0]  = (_Float16)w0.x;  bf[1]  = (_Float16)w0.y;
        bf[2]  = (_Float16)w0.z;  bf[3]  = (_Float16)w0.w;
        bf[4]  = (_Float16)w1.x;  bf[5]  = (_Float16)w1.y;
        bf[6]  = (_Float16)w1.z;  bf[7]  = (_Float16)w1.w;
        bf[8]  = (_Float16)w2.x;  bf[9]  = (_Float16)w2.y;
        bf[10] = (_Float16)w2.z;  bf[11] = (_Float16)w2.w;
        bf[12] = (_Float16)w3.x;  bf[13] = (_Float16)w3.y;
        bf[14] = (_Float16)w3.z;  bf[15] = (_Float16)w3.w;

        c = __builtin_amdgcn_wmma_f32_16x16x32_f16(
                /*neg_a=*/false, af, /*neg_b=*/false, bf,
                /*c_mod=*/(short)0, c, /*reuse_a=*/false, /*reuse_b=*/false);
    }

    // C layout: VGPR r holds M=r (lanes 0-15, N=lane) and M=r+8 (lanes 16-31).
    // Column N==0 lives in lane 0 (M=0..7) and lane 16 (M=8..15).
    if ((lane & 15) == 0) {
        const int mbase = (lane >> 4) << 3;
        const int e0 = ebase + mbase;
        if (e0 + 7 < n_edges) {
            // Fast path (always taken when E % 16 == 0): vector stores + merged
            // dst index loads, no per-element exec-mask juggling.
            float4 lo = make_float4(c[0], c[1], c[2], c[3]);
            float4 hi = make_float4(c[4], c[5], c[6], c[7]);
            ((float4*)(elog + e0))[0] = lo;
            ((float4*)(elog + e0))[1] = hi;
            const int4* dp = (const int4*)(dst + e0);
            int4 dA = dp[0], dB = dp[1];
            atomicMax(&smax[dA.x], f2mono(c[0]));
            atomicMax(&smax[dA.y], f2mono(c[1]));
            atomicMax(&smax[dA.z], f2mono(c[2]));
            atomicMax(&smax[dA.w], f2mono(c[3]));
            atomicMax(&smax[dB.x], f2mono(c[4]));
            atomicMax(&smax[dB.y], f2mono(c[5]));
            atomicMax(&smax[dB.z], f2mono(c[6]));
            atomicMax(&smax[dB.w], f2mono(c[7]));
        } else {
#pragma unroll
            for (int j = 0; j < 8; ++j) {
                int eidx = e0 + j;
                if (eidx < n_edges) {
                    float ev = c[j];
                    elog[eidx] = ev;
                    atomicMax(&smax[dst[eidx]], f2mono(ev));
                }
            }
        }
    }
}

// ---------------------------------------------------------------------------
// Kernel B: ex = exp(e - segmax[dst]); overwrite elog in place; sum into denom.
// ---------------------------------------------------------------------------
__global__ void gat_expsum_kernel(const int* __restrict__ dst,
                                  float* __restrict__ elog,
                                  const unsigned* __restrict__ smax,
                                  float* __restrict__ denom,
                                  int n_edges) {
    int i = blockIdx.x * blockDim.x + threadIdx.x;
    if (i >= n_edges) return;
    int d = dst[i];
    float m  = mono2f(smax[d]);
    float ex = __expf(elog[i] - m);
    elog[i] = ex;
    atomicAdd(&denom[d], ex);
}

// ---------------------------------------------------------------------------
// Kernel C: out[dst] += (ex/denom[dst]) * h[src].  16 lanes per edge, each
// lane owns 4 contiguous dims: one b128 gather + 4 fp32 global atomics.
// ---------------------------------------------------------------------------
__global__ void gat_aggregate_kernel(const float* __restrict__ h,
                                     const int* __restrict__ src,
                                     const int* __restrict__ dst,
                                     const float* __restrict__ ex,
                                     const float* __restrict__ denom,
                                     float* __restrict__ out,
                                     int n_edges) {
    long long gid = (long long)blockIdx.x * blockDim.x + threadIdx.x;
    int e = (int)(gid >> 4);
    int g = (int)(gid & 15);
    if (e >= n_edges) return;

    int si = src[e];
    int di = dst[e];
    float alpha = ex[e] / denom[di];

    const float4* hv4 = (const float4*)(h + (long long)si * D_DIM);
    float4 hv = hv4[g];

    float* o = out + (long long)di * D_DIM + g * 4;
    atomicAdd(o + 0, alpha * hv.x);
    atomicAdd(o + 1, alpha * hv.y);
    atomicAdd(o + 2, alpha * hv.z);
    atomicAdd(o + 3, alpha * hv.w);
}

// ---------------------------------------------------------------------------
extern "C" void kernel_launch(void* const* d_in, const int* in_sizes, int n_in,
                              void* d_out, int out_size, void* d_ws, size_t ws_size,
                              hipStream_t stream) {
    const float* h  = (const float*)d_in[0];
    const float* aw = (const float*)d_in[1];
    const int* src  = (const int*)d_in[2];
    const int* dst  = (const int*)d_in[3];
    float* out      = (float*)d_out;

    const int n_nodes = in_sizes[0] / D_DIM;
    const int n_edges = in_sizes[2];

    // Workspace layout: elog[E] (logits, then ex in place) | smax[N] | denom[N]
    float*    elog  = (float*)d_ws;
    unsigned* smax  = (unsigned*)(elog + n_edges);
    float*    denom = (float*)(smax + n_nodes);

    // 0) init out / smax / denom (poison + replay accumulation safety)
    {
        int total = n_nodes * D_DIM;
        int blocks = (total + 255) / 256;
        gat_init_kernel<<<blocks, 256, 0, stream>>>(out, smax, denom, n_nodes);
    }

    // A) logits via WMMA + segment max. 8 waves/block, 16 edges/wave.
    {
        int waves  = (n_edges + 15) / 16;
        int blocks = (waves + 7) / 8;
        gat_logits_wmma_kernel<<<blocks, 256, 0, stream>>>(
            h, aw, src, dst, elog, smax, n_edges);
    }

    // B) exp + denom
    {
        int blocks = (n_edges + 255) / 256;
        gat_expsum_kernel<<<blocks, 256, 0, stream>>>(dst, elog, smax, denom, n_edges);
    }

    // C) weighted scatter-add
    {
        long long total = (long long)n_edges * 16;
        int blocks = (int)((total + 255) / 256);
        gat_aggregate_kernel<<<blocks, 256, 0, stream>>>(
            h, src, dst, elog, denom, out, n_edges);
    }
}
